// PointTransformer_29085518528923
// MI455X (gfx1250) — matrix operand order repaired and checked by reading, not compiled, not run on hardware
//
#include <hip/hip_runtime.h>
#include <hip/hip_bf16.h>
#include <math.h>

#define NB        8
#define NPTS      2048
#define DIMC      32
#define KNN       16
#define POSI      64
#define VALI      4

typedef __attribute__((ext_vector_type(16))) _Float16 v16h;
typedef __attribute__((ext_vector_type(8)))  _Float16 v8h;
typedef __attribute__((ext_vector_type(8)))  float    v8f;

__device__ __forceinline__ v8f wmma16x16x32(v16h a, v16h b, v8f c) {
  // D = A(16x32 f16) * B(32x16 f16) + C(16x16 f32)
  return __builtin_amdgcn_wmma_f32_16x16x32_f16(false, a, false, b, (short)0, c,
                                                false, false);
}

// K index held by A-tile vector element e for lane-half kg (ISA 7.12.2, 16-bit A 16x32):
// elements 0..7 -> K = kg*8 + e ; elements 8..15 -> K = 16 + kg*8 + (e-8)
__device__ __forceinline__ int a_kidx(int e, int kg) {
  return (e & 7) + kg * 8 + ((e >= 8) ? 16 : 0);
}

// Build a 32x16 B-tile register image from row-major f32 W (zero-padded OOB).
// B layout: lane n=l&15 holds column n; lane-half kh holds K = a_kidx(e, kh).
__device__ __forceinline__ v16h load_b_tile(const float* __restrict__ W,
                                            int rows, int cols, int r0, int c0) {
  const int lane = threadIdx.x & 31;
  const int n  = lane & 15;
  const int kh = lane >> 4;
  v16h b;
#pragma unroll
  for (int e = 0; e < 16; ++e) {
    const int r = r0 + a_kidx(e, kh);
    const int c = c0 + n;
    float v = (r < rows && c < cols) ? W[r * cols + c] : 0.0f;
    b[e] = (_Float16)v;
  }
  return b;
}

// low 32 bits of a generic pointer to __shared__ == LDS byte offset
__device__ __forceinline__ unsigned lds_offset_of(const void* gen) {
  return (unsigned)(uintptr_t)gen;
}

// ---------------------------------------------------------------------------
// Prep kernel A: pack all 17 WMMA B-tiles (f16, exact per-lane register image)
// so hot kernels load each tile with one 32-byte vector load.
// Tile ids: 0-5 {A,B,C}w x 2 ncols ; 6-9 pm1 ; 10-13 pm2[kc][nc] ; 14 vm1 ; 15-16 vm2
// ---------------------------------------------------------------------------
__global__ __launch_bounds__(32) void pack_w_kernel(
    const float* __restrict__ Aw, const float* __restrict__ Bw,
    const float* __restrict__ Cw,
    const float* __restrict__ pm1w, const float* __restrict__ pm2w,
    const float* __restrict__ vm1w, const float* __restrict__ vm2w,
    _Float16* __restrict__ blob) {
  const int t = blockIdx.x;
  const float* W; int rows, cols, r0, c0;
  if (t < 6)        { const float* ws[3] = {Aw, Bw, Cw};
                      W = ws[t >> 1]; rows = DIMC; cols = DIMC; r0 = 0; c0 = (t & 1) * 16; }
  else if (t < 10)  { W = pm1w; rows = 3;    cols = POSI; r0 = 0;                 c0 = (t - 6) * 16; }
  else if (t < 14)  { W = pm2w; rows = POSI; cols = DIMC; r0 = ((t - 10) >> 1) * 32; c0 = ((t - 10) & 1) * 16; }
  else if (t == 14) { W = vm1w; rows = DIMC; cols = VALI; r0 = 0; c0 = 0; }
  else              { W = vm2w; rows = VALI; cols = DIMC; r0 = 0; c0 = (t - 15) * 16; }
  v16h v = load_b_tile(W, rows, cols, r0, c0);
  *(v16h*)(blob + (size_t)t * 512 + (threadIdx.x & 31) * 16) = v;
}

__device__ __forceinline__ v16h load_packed_tile(const _Float16* __restrict__ blob,
                                                 int tile) {
  return *(const v16h*)(blob + (size_t)tile * 512 + (threadIdx.x & 31) * 16);
}

// Prep kernel B: x f32 -> f16 (single memory-bound pass)
__global__ void cvt_x_kernel(const float* __restrict__ x,
                             _Float16* __restrict__ xh, int n) {
  const int t = blockIdx.x * blockDim.x + threadIdx.x;
  if (t < n) xh[t] = (_Float16)x[t];
}

// ---------------------------------------------------------------------------
// Kernel 1: streaming KNN. p[batch] staged into LDS on the CDNA5 async path
// (global_load_async_to_lds_b128, ASYNCcnt), one thread per query point with a
// 17-entry sorted top-k in registers; the 2048x2048 distance matrix is never
// materialized.
// ---------------------------------------------------------------------------
__global__ __launch_bounds__(256) void knn_kernel(const float* __restrict__ p,
                                                  int* __restrict__ idx_out) {
  __shared__ float sp[NPTS * 3];
  const int b = blockIdx.x / (NPTS / 256);
  const int i = (blockIdx.x % (NPTS / 256)) * 256 + threadIdx.x;
  const float* pb = p + (size_t)b * NPTS * 3;

  // async-copy 24 KB of positions into LDS: 1536 x 16B chunks, GVS addressing
  for (int t = threadIdx.x; t < (NPTS * 3) / 4; t += 256) {
    const unsigned lds = lds_offset_of(&sp[t * 4]);
    const unsigned go  = (unsigned)t * 16u;
    asm volatile("global_load_async_to_lds_b128 %0, %1, %2"
                 :: "v"(lds), "v"(go), "s"(pb) : "memory");
  }
  asm volatile("s_wait_asynccnt 0" ::: "memory");
  __syncthreads();

  const float qx = sp[i * 3 + 0], qy = sp[i * 3 + 1], qz = sp[i * 3 + 2];
  float bestd[KNN + 1];
  int   besti[KNN + 1];
#pragma unroll
  for (int t = 0; t <= KNN; ++t) { bestd[t] = 3.4e38f; besti[t] = 0; }

  for (int j = 0; j < NPTS; ++j) {
    const float dx = qx - sp[j * 3 + 0];
    const float dy = qy - sp[j * 3 + 1];
    const float dz = qz - sp[j * 3 + 2];
    const float d  = dx * dx + dy * dy + dz * dz;
    if (d < bestd[KNN]) {               // common path: one compare, no insert
      bestd[KNN] = d; besti[KNN] = j;
#pragma unroll
      for (int t = KNN; t > 0; --t) {
        if (bestd[t] < bestd[t - 1]) {  // strict < keeps jax-stable tie order
          float td = bestd[t]; bestd[t] = bestd[t - 1]; bestd[t - 1] = td;
          int   ti = besti[t]; besti[t] = besti[t - 1]; besti[t - 1] = ti;
        }
      }
    }
  }
  int* row = idx_out + ((size_t)b * NPTS + i) * KNN;
#pragma unroll
  for (int t = 0; t < KNN; ++t) row[t] = besti[t + 1];   // drop self (rank 0)
}

// ---------------------------------------------------------------------------
// Kernel 2: x@{A,B,C}_w + bias via WMMA. One wave per 16-row tile; A-tile is
// two b128 loads from pre-converted f16 x; B-tiles are packed register images.
// ---------------------------------------------------------------------------
__global__ __launch_bounds__(256) void proj_kernel(
    const _Float16* __restrict__ xh,
    const _Float16* __restrict__ blob,
    const float* __restrict__ Ab, const float* __restrict__ Bb,
    const float* __restrict__ Cb,
    float* __restrict__ xA, float* __restrict__ xB, float* __restrict__ xC) {
  const int lane = threadIdx.x & 31;
  const int wave = threadIdx.x >> 5;
  const int row0 = (blockIdx.x * 8 + wave) * 16;
  const int m  = lane & 15;
  const int kg = lane >> 4;

  // A-tile: elements 0..7 = K kg*8.., elements 8..15 = K 16+kg*8.. (contiguous 16B each)
  const _Float16* xr = xh + (size_t)(row0 + m) * DIMC;
  const v8h lo = *(const v8h*)(xr + kg * 8);
  const v8h hi = *(const v8h*)(xr + 16 + kg * 8);
  v16h a;
#pragma unroll
  for (int e = 0; e < 8; ++e) { a[e] = lo[e]; a[8 + e] = hi[e]; }

  const float* bs[3] = {Ab, Bb, Cb};
  float* outs[3] = {xA, xB, xC};
#pragma unroll
  for (int w = 0; w < 3; ++w) {
#pragma unroll
    for (int nc = 0; nc < 2; ++nc) {
      v16h bt = load_packed_tile(blob, w * 2 + nc);
      v8f c = {};
      c = wmma16x16x32(a, bt, c);
      const int col  = nc * 16 + m;
      const float bb = bs[w][col];
#pragma unroll
      for (int r = 0; r < 8; ++r)
        outs[w][(size_t)(row0 + kg * 8 + r) * DIMC + col] = c[r] + bb;
    }
  }
}

// ---------------------------------------------------------------------------
// Kernel 3: fused gather + pos-MLP + value-MLP + softmax + weighted sum.
// One wave per point: lane&15 = channel column, lane>>4 + VGPR index =
// neighbor row (C/D layout) -> softmax over neighbors is 8 in-register
// elements + one shfl_xor(16). 11 v_wmma per point.
// ---------------------------------------------------------------------------
__global__ __launch_bounds__(256) void attn_kernel(
    const float* __restrict__ p, const int* __restrict__ idx,
    const float* __restrict__ xA, const float* __restrict__ xB,
    const float* __restrict__ xC,
    const _Float16* __restrict__ blob,
    const float* __restrict__ pm1b, const float* __restrict__ pm2b,
    const float* __restrict__ vm1b, const float* __restrict__ vm2b,
    float* __restrict__ y) {
  __shared__ _Float16 stage_all[8][16 * POSI];   // 2 KB per-wave transpose buffer
  const int lane = threadIdx.x & 31;
  const int wave = threadIdx.x >> 5;
  _Float16* stage = stage_all[wave];
  const int gwave = blockIdx.x * 8 + wave;       // one point per wave
  const int b = gwave / NPTS;
  const int i = gwave - b * NPTS;
  const int m  = lane & 15;                      // A-tile row / D-tile column
  const int kg = lane >> 4;                      // lane half

  // Weight B-tiles: one 32-byte vector load each from the packed blob
  v16h B_pm1[4], B_pm2[2][2], B_vm1, B_vm2[2];
#pragma unroll
  for (int nc = 0; nc < 4; ++nc) B_pm1[nc] = load_packed_tile(blob, 6 + nc);
#pragma unroll
  for (int kc = 0; kc < 2; ++kc)
#pragma unroll
    for (int nc = 0; nc < 2; ++nc)
      B_pm2[kc][nc] = load_packed_tile(blob, 10 + kc * 2 + nc);
  B_vm1 = load_packed_tile(blob, 14);
#pragma unroll
  for (int nc = 0; nc < 2; ++nc) B_vm2[nc] = load_packed_tile(blob, 15 + nc);

  const size_t base = (size_t)b * NPTS;
  const int* nbr = idx + (size_t)gwave * KNN;

  // A-tile of position differences: row m = neighbor m, K=0..2 live (rest 0)
  const float pix = p[(base + i) * 3 + 0];
  const float piy = p[(base + i) * 3 + 1];
  const float piz = p[(base + i) * 3 + 2];
  const int jm = nbr[m];
  v16h a_pd = {};
  if (kg == 0) {
    a_pd[0] = (_Float16)(pix - p[(base + jm) * 3 + 0]);
    a_pd[1] = (_Float16)(piy - p[(base + jm) * 3 + 1]);
    a_pd[2] = (_Float16)(piz - p[(base + jm) * 3 + 2]);
  }

  // u = relu(pd @ pm1 + b1): 16x64, staged to LDS row-major f16
#pragma unroll
  for (int nc = 0; nc < 4; ++nc) {
    v8f c = {};
    c = wmma16x16x32(a_pd, B_pm1[nc], c);
    const int ucol = nc * 16 + m;
    const float bb = pm1b[ucol];
#pragma unroll
    for (int r = 0; r < 8; ++r) {
      float u = c[r] + bb;
      stage[(kg * 8 + r) * POSI + ucol] = (_Float16)(u > 0.f ? u : 0.f);
    }
  }
  asm volatile("s_wait_dscnt 0" ::: "memory");   // same-wave LDS transpose fence

  v16h a_u0, a_u1;                                // u as two 16x32 A-tiles
#pragma unroll
  for (int e = 0; e < 16; ++e) {
    a_u0[e] = stage[m * POSI + a_kidx(e, kg)];
    a_u1[e] = stage[m * POSI + 32 + a_kidx(e, kg)];
  }

  // pdv = u @ pm2 + b2 ; h = x_i - x_j1 + pdv ; val = pdv + x_j2
  v8f hmat[2], val[2];
#pragma unroll
  for (int nc = 0; nc < 2; ++nc) {
    v8f c = {};
    c = wmma16x16x32(a_u0, B_pm2[0][nc], c);
    c = wmma16x16x32(a_u1, B_pm2[1][nc], c);
    const int col = nc * 16 + m;
    const float bb = pm2b[col];
    const float xi = xA[(base + i) * DIMC + col];
#pragma unroll
    for (int r = 0; r < 8; ++r) {
      const int j = nbr[kg * 8 + r];
      const float pv = c[r] + bb;
      hmat[nc][r] = xi - xB[(base + j) * DIMC + col] + pv;
      val[nc][r]  = pv + xC[(base + j) * DIMC + col];
    }
  }

  // transpose h -> A-tile via LDS
#pragma unroll
  for (int nc = 0; nc < 2; ++nc) {
    const int col = nc * 16 + m;
#pragma unroll
    for (int r = 0; r < 8; ++r)
      stage[(kg * 8 + r) * DIMC + col] = (_Float16)hmat[nc][r];
  }
  asm volatile("s_wait_dscnt 0" ::: "memory");
  v16h a_h;
#pragma unroll
  for (int e = 0; e < 16; ++e) a_h[e] = stage[m * DIMC + a_kidx(e, kg)];

  // t = relu(h @ vm1 + vb1): only columns 0..3 live (vm1 zero-padded to 32x16)
  v8f tc = {};
  tc = wmma16x16x32(a_h, B_vm1, tc);
  if (m < VALI) {
    const float bb = vm1b[m];
#pragma unroll
    for (int r = 0; r < 8; ++r) {
      float t = tc[r] + bb;
      stage[(kg * 8 + r) * VALI + m] = (_Float16)(t > 0.f ? t : 0.f);
    }
  }
  asm volatile("s_wait_dscnt 0" ::: "memory");
  v16h a_t = {};                                  // 16x32 A-tile, K=0..3 live
  if (kg == 0) {
#pragma unroll
    for (int e = 0; e < VALI; ++e) a_t[e] = stage[m * VALI + e];
  }

  // hv = relu(t) @ vm2 + vb2 (16x32, D layout)
  v8f hv[2];
#pragma unroll
  for (int nc = 0; nc < 2; ++nc) {
    v8f c = {};
    c = wmma16x16x32(a_t, B_vm2[nc], c);
    const float bb = vm2b[nc * 16 + m];
#pragma unroll
    for (int r = 0; r < 8; ++r) hv[nc][r] = c[r] + bb;
  }

  // softmax over the 16 neighbor-rows per channel, then weighted value sum
#pragma unroll
  for (int nc = 0; nc < 2; ++nc) {
    float mx = -3.4e38f;
#pragma unroll
    for (int r = 0; r < 8; ++r) mx = fmaxf(mx, hv[nc][r]);
    mx = fmaxf(mx, __shfl_xor(mx, 16, 32));
    float e8[8], sum = 0.f;
#pragma unroll
    for (int r = 0; r < 8; ++r) { e8[r] = __expf(hv[nc][r] - mx); sum += e8[r]; }
    sum += __shfl_xor(sum, 16, 32);
    const float inv = 1.0f / sum;
    float acc = 0.f;
#pragma unroll
    for (int r = 0; r < 8; ++r) acc += e8[r] * inv * val[nc][r];
    acc += __shfl_xor(acc, 16, 32);
    if (kg == 0) y[(size_t)gwave * DIMC + nc * 16 + m] = acc;
  }
}

__global__ void copy_p_kernel(const float* __restrict__ p, float* __restrict__ o) {
  const int t = blockIdx.x * blockDim.x + threadIdx.x;
  if (t < NB * NPTS * 3) o[t] = p[t];
}

extern "C" void kernel_launch(void* const* d_in, const int* in_sizes, int n_in,
                              void* d_out, int out_size, void* d_ws, size_t ws_size,
                              hipStream_t stream) {
  const float* x    = (const float*)d_in[0];
  const float* p    = (const float*)d_in[1];
  const float* Aw   = (const float*)d_in[2];
  const float* Ab   = (const float*)d_in[3];
  const float* Bw   = (const float*)d_in[4];
  const float* Bb   = (const float*)d_in[5];
  const float* Cw   = (const float*)d_in[6];
  const float* Cb   = (const float*)d_in[7];
  const float* pm1w = (const float*)d_in[8];
  const float* pm1b = (const float*)d_in[9];
  const float* pm2w = (const float*)d_in[10];
  const float* pm2b = (const float*)d_in[11];
  const float* vm1w = (const float*)d_in[12];
  const float* vm1b = (const float*)d_in[13];
  const float* vm2w = (const float*)d_in[14];
  const float* vm2b = (const float*)d_in[15];

  // workspace layout (bytes):
  //   [0      , 1 MB) : idx  (16384*16 i32)
  //   [1 MB   , 3 MB) : xA   (16384*32 f32)
  //   [3 MB   , 5 MB) : xB
  //   [5 MB   , 7 MB) : xC
  //   [7 MB   , 8 MB) : xh   (16384*32 f16)
  //   [8 MB   , ...)  : packed weight B-tiles (17 * 1 KB)
  const size_t MB = 1u << 20;
  int*      idx  = (int*)d_ws;
  float*    xA   = (float*)((char*)d_ws + 1 * MB);
  float*    xB   = (float*)((char*)d_ws + 3 * MB);
  float*    xC   = (float*)((char*)d_ws + 5 * MB);
  _Float16* xh   = (_Float16*)((char*)d_ws + 7 * MB);
  _Float16* blob = (_Float16*)((char*)d_ws + 8 * MB);

  float* y    = (float*)d_out;
  float* outp = y + (size_t)NB * NPTS * DIMC;

  const int nx = NB * NPTS * DIMC;
  pack_w_kernel<<<17, 32, 0, stream>>>(Aw, Bw, Cw, pm1w, pm2w, vm1w, vm2w, blob);
  cvt_x_kernel<<<(nx + 255) / 256, 256, 0, stream>>>(x, xh, nx);
  knn_kernel<<<NB * (NPTS / 256), 256, 0, stream>>>(p, idx);
  proj_kernel<<<(NB * NPTS) / 128, 256, 0, stream>>>(xh, blob, Ab, Bb, Cb,
                                                     xA, xB, xC);
  attn_kernel<<<(NB * NPTS) / 8, 256, 0, stream>>>(p, idx, xA, xB, xC, blob,
                                                   pm1b, pm2b, vm1b, vm2b, y);
  copy_p_kernel<<<(NB * NPTS * 3 + 255) / 256, 256, 0, stream>>>(p, outp);
}
